// Model_33552284516393
// MI455X (gfx1250) — compile-verified, matrix operand order checked by reading, not run on hardware
//
#include <hip/hip_runtime.h>
#include <hip/hip_bf16.h>

// ---------------------------------------------------------------------------
// GIN inference pipeline for MI455X (gfx1250, wave32, WMMA).
// GEMMs run on v_wmma_f32_16x16x32_bf16 (f32 accum, bf16 operands rounded (RNE)
// from f32 inputs). Aggregation is moved after the first MLP linear (legal:
// segment-sum commutes with the linear map), shrinking layer-0 scatter 8x.
// ---------------------------------------------------------------------------

typedef __attribute__((ext_vector_type(16))) __bf16 v16bf;
typedef __attribute__((ext_vector_type(8)))  float  v8f;

static constexpr int    Nn = 4096;   // nodes == in_feats
static constexpr int    Hh = 512;    // hidden
static constexpr int    Ee = 65536;  // edges
static constexpr size_t NH = (size_t)Nn * Hh;

__device__ __forceinline__ unsigned short f2bf(float f) {
  union { float f; unsigned int u; } cv; cv.f = f;
  unsigned int u = cv.u;
  return (unsigned short)((u + 0x7FFFu + ((u >> 16) & 1u)) >> 16); // RNE
}

// ---------------------------------------------------------------------------
// Tiled bf16 WMMA GEMM: C[M=4096, N=512] = A[M,K] * B[K,512]  (all f32 in mem)
// Block = 256 threads = 8 waves; block tile 128x128; wave tile 32x64; BK = 32.
// ---------------------------------------------------------------------------
__global__ __launch_bounds__(256) void gemm_bf16_wmma(
    const float* __restrict__ A, const float* __restrict__ B,
    float* __restrict__ C, int K) {
  constexpr int TM = 128, TN = 128, TK = 32, PAD = 8;
  __shared__ __align__(16) unsigned short lA[TM][TK + PAD]; // 80B rows, 16B-aligned
  __shared__ __align__(16) unsigned short lB[TN][TK + PAD]; // stored transposed [n][k]

  const int tid  = threadIdx.x;
  const int wave = tid >> 5;       // 0..7
  const int lane = tid & 31;
  const int wm   = wave >> 1;      // 0..3 : 32-row strip
  const int wn   = wave & 1;       // 0..1 : 64-col strip
  const int half = lane >> 4;      // K-half select (ISA 16-bit A/B lane layout)
  const int l16  = lane & 15;

  const int bm = blockIdx.y * TM;
  const int bn = blockIdx.x * TN;

  v8f acc[2][4];
#pragma unroll
  for (int i = 0; i < 2; ++i)
#pragma unroll
    for (int j = 0; j < 4; ++j) acc[i][j] = (v8f)0.0f;

  const int kIters = K / TK;
  for (int kb = 0; kb < kIters; ++kb) {
    // ---- stage A tile: 128x32 f32 -> bf16 (each thread: 4 x float4) ----
#pragma unroll
    for (int i = 0; i < 4; ++i) {
      int e   = tid + i * 256;     // float4 index 0..1023
      int row = e >> 3;            // 8 float4 per 32-float row
      int c4  = e & 7;
      const float4 f = *(const float4*)(A + (size_t)(bm + row) * K + kb * TK + c4 * 4);
      unsigned long long p =  (unsigned long long)f2bf(f.x)
                           | ((unsigned long long)f2bf(f.y) << 16)
                           | ((unsigned long long)f2bf(f.z) << 32)
                           | ((unsigned long long)f2bf(f.w) << 48);
      *(unsigned long long*)&lA[row][c4 * 4] = p;
    }
    // ---- stage B tile transposed: lB[n][k] = B[(kb*32+k)*512 + bn + n] ----
#pragma unroll
    for (int i = 0; i < 4; ++i) {
      int e  = tid + i * 256;      // float4 index over 32(k) x 32(n/4)
      int kk = e >> 5;
      int n4 = e & 31;
      const float4 f = *(const float4*)(B + (size_t)(kb * TK + kk) * Hh + bn + n4 * 4);
      lB[n4 * 4 + 0][kk] = f2bf(f.x);
      lB[n4 * 4 + 1][kk] = f2bf(f.y);
      lB[n4 * 4 + 2][kk] = f2bf(f.z);
      lB[n4 * 4 + 3][kk] = f2bf(f.w);
    }
    if (kb + 1 < kIters)  // pull next A tile toward L2/WGP$
      __builtin_prefetch(A + (size_t)(bm + (tid >> 1)) * K + (kb + 1) * TK, 0, 0);
    __syncthreads();

    // ---- operand fragments (two ds_load_b128 each, per ISA lane layout:
    //      lanes 0-15: K 0-7 & 16-23, lanes 16-31: K 8-15 & 24-31) ----
    union FragU { v16bf v; uint4 q[2]; } aF[2], bF[4];
#pragma unroll
    for (int i = 0; i < 2; ++i) {
      int row = wm * 32 + i * 16 + l16;
      aF[i].q[0] = *(const uint4*)&lA[row][half * 8];
      aF[i].q[1] = *(const uint4*)&lA[row][16 + half * 8];
    }
#pragma unroll
    for (int j = 0; j < 4; ++j) {
      int col = wn * 64 + j * 16 + l16;
      bF[j].q[0] = *(const uint4*)&lB[col][half * 8];
      bF[j].q[1] = *(const uint4*)&lB[col][16 + half * 8];
    }
#pragma unroll
    for (int i = 0; i < 2; ++i)
#pragma unroll
      for (int j = 0; j < 4; ++j)
        acc[i][j] = __builtin_amdgcn_wmma_f32_16x16x32_bf16(
            false, aF[i].v, false, bF[j].v, (short)0, acc[i][j], false, false);
    __syncthreads();
  }

  // D layout: VGPR r -> lanes 0-15: M=r, N=l16 ; lanes 16-31: M=r+8, N=l16
#pragma unroll
  for (int i = 0; i < 2; ++i)
#pragma unroll
    for (int j = 0; j < 4; ++j) {
      int n = bn + wn * 64 + j * 16 + l16;
#pragma unroll
      for (int r = 0; r < 8; ++r) {
        int m = bm + wm * 32 + i * 16 + half * 8 + r;
        C[(size_t)m * Hh + n] = acc[i][j][r];
      }
    }
}

// ---------------------------------------------------------------------------
// zero a buffer (float4 granularity)
// ---------------------------------------------------------------------------
__global__ __launch_bounds__(256) void zero_f4(float* __restrict__ p, int n4) {
  int i = blockIdx.x * blockDim.x + threadIdx.x;
  if (i < n4) ((float4*)p)[i] = make_float4(0.f, 0.f, 0.f, 0.f);
}

// ---------------------------------------------------------------------------
// edge scatter-add in H-space: agg[dst] += y[src]   (one block per edge)
// ---------------------------------------------------------------------------
__global__ __launch_bounds__(256) void edge_agg(
    const float* __restrict__ y, const int* __restrict__ src,
    const int* __restrict__ dst, float* __restrict__ agg) {
  int e = blockIdx.x;
  const float* yr = y   + (size_t)src[e] * Hh;
  float*       ar = agg + (size_t)dst[e] * Hh;
  int t = threadIdx.x;
  atomicAdd(&ar[t],       yr[t]);
  atomicAdd(&ar[t + 256], yr[t + 256]);
}

// ---------------------------------------------------------------------------
// x = (1+eps)*y + agg + ba ; then BN(eval) ; ReLU
// ---------------------------------------------------------------------------
__global__ __launch_bounds__(256) void fuse_pre(
    const float* __restrict__ y, const float* __restrict__ agg,
    const float* __restrict__ eps_p, const float* __restrict__ ba,
    const float* __restrict__ g, const float* __restrict__ b,
    const float* __restrict__ m, const float* __restrict__ v,
    float* __restrict__ out) {
  size_t i = (size_t)blockIdx.x * blockDim.x + threadIdx.x;
  int c = (int)(i & (Hh - 1));
  float e1 = 1.0f + *eps_p;
  float x = fmaf(e1, y[i], agg[i]) + ba[c];
  float t = fmaf((x - m[c]) * rsqrtf(v[c] + 1e-5f), g[c], b[c]);
  out[i] = fmaxf(t, 0.0f);
}

// ---------------------------------------------------------------------------
// h = ReLU(BN_O(ReLU(BN_A(x + bias))))   (in-place capable)
// ---------------------------------------------------------------------------
__global__ __launch_bounds__(256) void fuse_post(
    const float* __restrict__ x, const float* __restrict__ bias,
    const float* __restrict__ ga, const float* __restrict__ ba,
    const float* __restrict__ ma, const float* __restrict__ va,
    const float* __restrict__ go, const float* __restrict__ bo,
    const float* __restrict__ mo, const float* __restrict__ vo,
    float* __restrict__ out) {
  size_t i = (size_t)blockIdx.x * blockDim.x + threadIdx.x;
  int c = (int)(i & (Hh - 1));
  float t = x[i] + bias[c];
  t = fmaxf(fmaf((t - ma[c]) * rsqrtf(va[c] + 1e-5f), ga[c], ba[c]), 0.0f);
  t = fmaxf(fmaf((t - mo[c]) * rsqrtf(vo[c] + 1e-5f), go[c], bo[c]), 0.0f);
  out[i] = t;
}

// ---------------------------------------------------------------------------
// readout: partial dot of x[n] with 2-column matrix w[n,2]; one partial/block
// ---------------------------------------------------------------------------
__global__ __launch_bounds__(256) void dot2_partial(
    const float* __restrict__ x, const float* __restrict__ w,
    size_t n, float* __restrict__ part) {
  float s0 = 0.f, s1 = 0.f;
  size_t stride = (size_t)gridDim.x * blockDim.x;
  for (size_t i = (size_t)blockIdx.x * blockDim.x + threadIdx.x; i < n; i += stride) {
    float  xv = x[i];
    float2 wv = *(const float2*)(w + (i << 1));
    s0 = fmaf(xv, wv.x, s0);
    s1 = fmaf(xv, wv.y, s1);
  }
#pragma unroll
  for (int off = 16; off > 0; off >>= 1) {
    s0 += __shfl_down(s0, off, 32);
    s1 += __shfl_down(s1, off, 32);
  }
  __shared__ float r0[8], r1[8];
  int lane = threadIdx.x & 31, wv = threadIdx.x >> 5;
  if (lane == 0) { r0[wv] = s0; r1[wv] = s1; }
  __syncthreads();
  if (threadIdx.x == 0) {
    float t0 = 0.f, t1 = 0.f;
#pragma unroll
    for (int i = 0; i < 8; ++i) { t0 += r0[i]; t1 += r1[i]; }
    part[2 * blockIdx.x + 0] = t0;
    part[2 * blockIdx.x + 1] = t1;
  }
}

// deterministic final combine: out[c] = sum(partials) + bp0[c] + sum_i bpR[i][c]
__global__ __launch_bounds__(256) void final_reduce(
    const float* __restrict__ part, int nent,
    const float* __restrict__ bp0, const float* __restrict__ bpR,
    float* __restrict__ out) {
  float s0 = 0.f, s1 = 0.f;
  for (int i = threadIdx.x; i < nent; i += 256) {
    s0 += part[2 * i + 0];
    s1 += part[2 * i + 1];
  }
#pragma unroll
  for (int off = 16; off > 0; off >>= 1) {
    s0 += __shfl_down(s0, off, 32);
    s1 += __shfl_down(s1, off, 32);
  }
  __shared__ float r0[8], r1[8];
  int lane = threadIdx.x & 31, wv = threadIdx.x >> 5;
  if (lane == 0) { r0[wv] = s0; r1[wv] = s1; }
  __syncthreads();
  if (threadIdx.x == 0) {
    float t0 = bp0[0], t1 = bp0[1];
#pragma unroll
    for (int i = 0; i < 4; ++i) { t0 += bpR[2 * i]; t1 += bpR[2 * i + 1]; }
#pragma unroll
    for (int i = 0; i < 8; ++i) { t0 += r0[i]; t1 += r1[i]; }
    out[0] = t0;
    out[1] = t1;
  }
}

// ---------------------------------------------------------------------------
extern "C" void kernel_launch(void* const* d_in, const int* in_sizes, int n_in,
                              void* d_out, int out_size, void* d_ws, size_t ws_size,
                              hipStream_t stream) {
  const float* feat = (const float*)d_in[0];
  const int*   esrc = (const int*)  d_in[1];
  const int*   edst = (const int*)  d_in[2];
  const float* eps0 = (const float*)d_in[3];
  const float* W0a  = (const float*)d_in[4];
  const float* b0a  = (const float*)d_in[5];
  const float* W0b  = (const float*)d_in[6];
  const float* b0b  = (const float*)d_in[7];
  const float* bn0a_g = (const float*)d_in[8],  *bn0a_b = (const float*)d_in[9];
  const float* bn0a_m = (const float*)d_in[10], *bn0a_v = (const float*)d_in[11];
  const float* bnA0_g = (const float*)d_in[12], *bnA0_b = (const float*)d_in[13];
  const float* bnA0_m = (const float*)d_in[14], *bnA0_v = (const float*)d_in[15];
  const float* bnO0_g = (const float*)d_in[16], *bnO0_b = (const float*)d_in[17];
  const float* bnO0_m = (const float*)d_in[18], *bnO0_v = (const float*)d_in[19];
  const float* epsR = (const float*)d_in[20];
  const float* WRa  = (const float*)d_in[21];
  const float* bRa  = (const float*)d_in[22];
  const float* WRb  = (const float*)d_in[23];
  const float* bRb  = (const float*)d_in[24];
  const float* bnRa_g = (const float*)d_in[25], *bnRa_b = (const float*)d_in[26];
  const float* bnRa_m = (const float*)d_in[27], *bnRa_v = (const float*)d_in[28];
  const float* bnAR_g = (const float*)d_in[29], *bnAR_b = (const float*)d_in[30];
  const float* bnAR_m = (const float*)d_in[31], *bnAR_v = (const float*)d_in[32];
  const float* bnOR_g = (const float*)d_in[33], *bnOR_b = (const float*)d_in[34];
  const float* bnOR_m = (const float*)d_in[35], *bnOR_v = (const float*)d_in[36];
  const float* Wp0 = (const float*)d_in[37];
  const float* bp0 = (const float*)d_in[38];
  const float* WpR = (const float*)d_in[39];
  const float* bpR = (const float*)d_in[40];
  float* out = (float*)d_out;

  // workspace: 4 hidden reps + Y + AGG + X1 + readout partials
  float* ws  = (float*)d_ws;
  float* hs0 = ws + 0 * NH;
  float* hs1 = ws + 1 * NH;
  float* hs2 = ws + 2 * NH;
  float* hs3 = ws + 3 * NH;
  float* Y   = ws + 4 * NH;
  float* AGG = ws + 5 * NH;
  float* X1  = ws + 6 * NH;
  float* PART = ws + 7 * NH;   // 6144 block partials * 2 floats
  float* hs[4] = {hs0, hs1, hs2, hs3};

  const dim3 gG(Hh / 128, Nn / 128), bG(256);   // (4, 32) x 256
  const int  nElemBlocks = (int)(NH / 256);     // 8192
  const int  nZeroBlocks = (int)(NH / 4 / 256); // 2048

  // ---------------- layer 0 (K = 4096) ----------------
  gemm_bf16_wmma<<<gG, bG, 0, stream>>>(feat, W0a, Y, Nn);
  zero_f4<<<nZeroBlocks, 256, 0, stream>>>(AGG, (int)(NH / 4));
  edge_agg<<<Ee, 256, 0, stream>>>(Y, esrc, edst, AGG);
  fuse_pre<<<nElemBlocks, 256, 0, stream>>>(Y, AGG, eps0, b0a,
                                            bn0a_g, bn0a_b, bn0a_m, bn0a_v, X1);
  gemm_bf16_wmma<<<gG, bG, 0, stream>>>(X1, W0b, hs0, Hh);
  fuse_post<<<nElemBlocks, 256, 0, stream>>>(hs0, b0b,
                                             bnA0_g, bnA0_b, bnA0_m, bnA0_v,
                                             bnO0_g, bnO0_b, bnO0_m, bnO0_v, hs0);

  // ---------------- stacked layers 1..3 (K = 512) ----------------
  for (int r = 0; r < 3; ++r) {
    const size_t wOff = (size_t)r * Hh * Hh;
    const size_t cOff = (size_t)r * Hh;
    gemm_bf16_wmma<<<gG, bG, 0, stream>>>(hs[r], WRa + wOff, Y, Hh);
    zero_f4<<<nZeroBlocks, 256, 0, stream>>>(AGG, (int)(NH / 4));
    edge_agg<<<Ee, 256, 0, stream>>>(Y, esrc, edst, AGG);
    fuse_pre<<<nElemBlocks, 256, 0, stream>>>(Y, AGG, epsR + r, bRa + cOff,
                                              bnRa_g + cOff, bnRa_b + cOff,
                                              bnRa_m + cOff, bnRa_v + cOff, X1);
    gemm_bf16_wmma<<<gG, bG, 0, stream>>>(X1, WRb + wOff, hs[r + 1], Hh);
    fuse_post<<<nElemBlocks, 256, 0, stream>>>(hs[r + 1], bRb + cOff,
                                               bnAR_g + cOff, bnAR_b + cOff,
                                               bnAR_m + cOff, bnAR_v + cOff,
                                               bnOR_g + cOff, bnOR_b + cOff,
                                               bnOR_m + cOff, bnOR_v + cOff,
                                               hs[r + 1]);
  }

  // ---------------- jumping-knowledge readout ----------------
  dot2_partial<<<2048, 256, 0, stream>>>(feat, Wp0, (size_t)Nn * Nn, PART);
  for (int i = 0; i < 4; ++i)
    dot2_partial<<<1024, 256, 0, stream>>>(hs[i], WpR + (size_t)i * NH * 2, NH,
                                           PART + (size_t)(2048 + i * 1024) * 2);
  final_reduce<<<1, 256, 0, stream>>>(PART, 2048 + 4 * 1024, bp0, bpR, out);
}
